// Stacked_Encoder_60576218743337
// MI455X (gfx1250) — matrix operand order repaired and checked by reading, not compiled, not run on hardware
//
#include <hip/hip_runtime.h>
#include <math.h>

// ---------------- constants (match reference) ----------------
#define LAY 6
#define ED  512
#define QD  64
#define VD  64
#define NH  8
#define FF  2048
#define BSZ 8
#define SLN 1024
#define NTOK (BSZ * SLN)   // 8192 tokens

typedef __bf16 bf16;
typedef __bf16 bf16x16 __attribute__((ext_vector_type(16)));
typedef __bf16 bf16x8  __attribute__((ext_vector_type(8)));
typedef float  f32x8   __attribute__((ext_vector_type(8)));

union FragU { bf16x16 v; bf16x8 h[2]; };

__device__ __forceinline__ f32x8 zero8() {
  f32x8 z;
#pragma unroll
  for (int i = 0; i < 8; i++) z[i] = 0.0f;
  return z;
}

__device__ __forceinline__ f32x8 wmma_bf16(bf16x16 a, bf16x16 b, f32x8 c) {
  return __builtin_amdgcn_wmma_f32_16x16x32_bf16(false, a, false, b,
                                                 (short)0, c, false, false);
}

// async global->LDS 128b copy (ISA §10: GLOBAL_LOAD_ASYNC_TO_LDS_B128, ASYNCcnt)
__device__ __forceinline__ void async_b128(unsigned lds_off, unsigned g_off,
                                           unsigned long long base) {
  asm volatile("global_load_async_to_lds_b128 %0, %1, %2"
               :: "v"(lds_off), "v"(g_off), "s"(base)
               : "memory");
}
__device__ __forceinline__ void wait_async0() {
  asm volatile("s_wait_asynccnt 0x0" ::: "memory");
}
__device__ __forceinline__ void wait_async4() {
  // previous tile's 4 copies (issued earlier, in-order) are done;
  // the 4 just-issued ones may still be in flight.
  asm volatile("s_wait_asynccnt 0x4" ::: "memory");
}
__device__ __forceinline__ unsigned lds_addr(const void* p) {
  // generic LDS pointer: low 32 bits are the LDS offset (flat aperture rule)
  return (unsigned)(size_t)p;
}

// ---------------- f32 -> bf16 convert ----------------
__global__ void cvt_f32_bf16(const float* __restrict__ in,
                             bf16* __restrict__ out, int n) {
  int i = blockIdx.x * blockDim.x + threadIdx.x;
  if (i < n) out[i] = (bf16)in[i];
}

// f32 (K x N) -> bf16 transposed (N x K)
__global__ void cvt_t(const float* __restrict__ in, bf16* __restrict__ out,
                      int K, int N) {
  int i = blockIdx.x * blockDim.x + threadIdx.x;
  if (i < K * N) {
    int n = i / K, k = i - n * K;
    out[i] = (bf16)in[(size_t)k * N + n];
  }
}

// ---------------- WMMA GEMM: C[M,N] = A[M,K] * Bt[N,K]^T + bias --------------
// A row-major (M x K) bf16; Bt row-major (N x K) bf16 (weights pre-transposed).
// 128x128 block, BK=32, 8 waves. Double-buffered async b128 tile staging.
template <int RELU>
__global__ __launch_bounds__(256) void gemm_bf16(
    const bf16* __restrict__ A, const bf16* __restrict__ Bt,
    const float* __restrict__ bias, float* __restrict__ Cf,
    bf16* __restrict__ Cb, int M, int N, int K) {
  __shared__ __align__(16) bf16 As[2][128][32];
  __shared__ __align__(16) bf16 Bs[2][128][32];   // [n][k]

  const int tid  = threadIdx.x;
  const int lane = tid & 31;
  const int w    = tid >> 5;
  const int m_off = (w & 3) * 32;
  const int n_off = (w >> 2) * 64;
  const int bm = blockIdx.y * 128;
  const int bn = blockIdx.x * 128;
  const int lg = lane >> 4;
  const int lr = lane & 15;
  const unsigned long long Abase = (unsigned long long)(size_t)A;
  const unsigned long long Bbase = (unsigned long long)(size_t)Bt;

  auto stage = [&](int buf, int kk) {
#pragma unroll
    for (int i = 0; i < 2; i++) {
      int s = tid + i * 256;
      int r = s >> 2, c8 = (s & 3) * 8;
      async_b128(lds_addr(&As[buf][r][c8]),
                 (unsigned)(((size_t)(bm + r) * K + kk + c8) * 2), Abase);
      async_b128(lds_addr(&Bs[buf][r][c8]),
                 (unsigned)(((size_t)(bn + r) * K + kk + c8) * 2), Bbase);
    }
  };

  f32x8 acc[2][4];
#pragma unroll
  for (int i = 0; i < 2; i++)
#pragma unroll
    for (int j = 0; j < 4; j++) acc[i][j] = zero8();

  const int nk = K / 32;
  stage(0, 0);
  for (int kt = 0; kt < nk; kt++) {
    if (kt + 1 < nk) {
      stage((kt + 1) & 1, (kt + 1) * 32);  // prefetch next tile
      wait_async4();                       // current tile landed
    } else {
      wait_async0();
    }
    __syncthreads();

    const int buf = kt & 1;
    FragU a[2], b[4];
#pragma unroll
    for (int ms = 0; ms < 2; ms++) {
      const bf16* rp = &As[buf][m_off + ms * 16 + lr][0];
      a[ms].h[0] = *(const bf16x8*)(rp + 8 * lg);        // K = j   + 8*lg
      a[ms].h[1] = *(const bf16x8*)(rp + 16 + 8 * lg);   // K = j+8 + 8*lg
    }
#pragma unroll
    for (int ns = 0; ns < 4; ns++) {
      const bf16* cp = &Bs[buf][n_off + ns * 16 + lr][0];
      b[ns].h[0] = *(const bf16x8*)(cp + 8 * lg);        // K = j   + 8*lg
      b[ns].h[1] = *(const bf16x8*)(cp + 16 + 8 * lg);
    }
#pragma unroll
    for (int ms = 0; ms < 2; ms++)
#pragma unroll
      for (int ns = 0; ns < 4; ns++)
        acc[ms][ns] = wmma_bf16(a[ms].v, b[ns].v, acc[ms][ns]);
    __syncthreads();  // all reads of this buffer done before it is restaged
  }

#pragma unroll
  for (int ms = 0; ms < 2; ms++)
#pragma unroll
    for (int ns = 0; ns < 4; ns++)
#pragma unroll
      for (int v = 0; v < 8; v++) {
        int gm = bm + m_off + ms * 16 + v + 8 * lg;
        int gn = bn + n_off + ns * 16 + lr;
        float val = acc[ms][ns][v] + bias[gn];
        if (RELU) val = val > 0.0f ? val : 0.0f;
        if (Cf) Cf[(size_t)gm * N + gn] = val;
        if (Cb) Cb[(size_t)gm * N + gn] = (bf16)val;
      }
}

// ---------------- attention: flash-style, WMMA for QK^T and PV ---------------
// grid (SL/64, NH, BS), 128 threads = 4 waves; wave handles 16 query rows.
// Head h feature d lives at column d*NH + h (reference reshape (sl,QD,NH)).
// O accumulator kept in registers (WMMA C layout), scaled by softmax factor.
__global__ __launch_bounds__(128) void attention_kernel(
    const bf16* __restrict__ Qg, const bf16* __restrict__ Kg,
    const bf16* __restrict__ Vg, const int* __restrict__ msk,
    bf16* __restrict__ Zg) {
  __shared__ __align__(16) bf16  Ks[64][64];       // [k-row][d]
  __shared__ __align__(16) bf16  Vt[64][64];       // [d][k-row] (transposed)
  __shared__ __align__(16) bf16  QPs[4][16][64];   // Q staging, then P (bf16)
  __shared__ float Ss[4][16][64];
  __shared__ float Os[4][16][64];
  __shared__ float rowm[4][16], rowl[4][16], rowf[4][16];
  __shared__ int   mks[64];

  const int tid  = threadIdx.x;
  const int lane = tid & 31;
  const int w    = tid >> 5;
  const int lg = lane >> 4, lr = lane & 15;
  const int qb = blockIdx.x * 64;
  const int h  = blockIdx.y;
  const int b  = blockIdx.z;

  for (int i = 0; i < 32; i++) {
    int flat = lane * 32 + i;
    int r = flat >> 6, d = flat & 63;
    QPs[w][r][d] = Qg[(size_t)(b * SLN + qb + w * 16 + r) * ED + d * NH + h];
  }
  if (lane < 16) { rowm[w][lane] = -3.0e38f; rowl[w][lane] = 0.0f; }
  __syncthreads();

  FragU qa[2];
  {
    const bf16* qr = &QPs[w][lr][0];
#pragma unroll
    for (int c = 0; c < 2; c++) {
      qa[c].h[0] = *(const bf16x8*)(qr + 32 * c + 8 * lg);
      qa[c].h[1] = *(const bf16x8*)(qr + 32 * c + 16 + 8 * lg);
    }
  }

  f32x8 oacc[4];
#pragma unroll
  for (int t = 0; t < 4; t++) oacc[t] = zero8();

  for (int kb = 0; kb < SLN; kb += 64) {
    __syncthreads();   // protect Ks/Vt/QPs vs previous iteration
    for (int i = 0; i < 32; i++) {            // cooperative K/V staging
      int flat = tid * 32 + i;
      int r = flat >> 6, d = flat & 63;
      Ks[r][d] = Kg[(size_t)(b * SLN + kb + r) * ED + d * NH + h];
      Vt[d][r] = Vg[(size_t)(b * SLN + kb + r) * ED + d * NH + h];
    }
    if (tid < 64) mks[tid] = msk[b * SLN + kb + tid];
    __syncthreads();

    // S(16x64) = Q * K^T
#pragma unroll
    for (int t = 0; t < 4; t++) {
      FragU kb2[2];
      const bf16* kr = &Ks[16 * t + lr][0];
#pragma unroll
      for (int c = 0; c < 2; c++) {
        kb2[c].h[0] = *(const bf16x8*)(kr + 32 * c + 8 * lg);
        kb2[c].h[1] = *(const bf16x8*)(kr + 32 * c + 16 + 8 * lg);
      }
      f32x8 s = zero8();
      s = wmma_bf16(qa[0].v, kb2[0].v, s);
      s = wmma_bf16(qa[1].v, kb2[1].v, s);
#pragma unroll
      for (int v = 0; v < 8; v++)
        Ss[w][v + 8 * lg][16 * t + lr] = s[v];
    }
    __syncthreads();

    // online softmax: mask (-1e9) BEFORE /sqrt(64), per reference; P -> bf16
    if (lane < 16) {
      int r = lane;
      float bm = -3.0e38f;
#pragma unroll 4
      for (int j = 0; j < 64; j++) {
        float sc = (mks[j] != 0 ? Ss[w][r][j] : -1.0e9f) * 0.125f;
        Ss[w][r][j] = sc;
        bm = fmaxf(bm, sc);
      }
      float om = rowm[w][r];
      float nm = fmaxf(om, bm);
      float f  = __expf(om - nm);
      float sum = 0.0f;
#pragma unroll 4
      for (int j = 0; j < 64; j++) {
        float pv = __expf(Ss[w][r][j] - nm);
        QPs[w][r][j] = (bf16)pv;     // P as bf16 for WMMA A operand
        sum += pv;
      }
      rowm[w][r] = nm;
      rowl[w][r] = rowl[w][r] * f + sum;
      rowf[w][r] = f;
    }
    __syncthreads();

    // O = O*f + P(16x64) * V(64x64)  — O stays in registers (C layout)
    FragU pa[2];
    {
      const bf16* pr = &QPs[w][lr][0];
#pragma unroll
      for (int c = 0; c < 2; c++) {
        pa[c].h[0] = *(const bf16x8*)(pr + 32 * c + 8 * lg);
        pa[c].h[1] = *(const bf16x8*)(pr + 32 * c + 16 + 8 * lg);
      }
    }
    float fr[8];
#pragma unroll
    for (int v = 0; v < 8; v++) fr[v] = rowf[w][v + 8 * lg];  // row M = v+8*lg
#pragma unroll
    for (int t = 0; t < 4; t++) {
      FragU vb[2];
      const bf16* vr = &Vt[16 * t + lr][0];
#pragma unroll
      for (int c = 0; c < 2; c++) {
        vb[c].h[0] = *(const bf16x8*)(vr + 32 * c + 8 * lg);
        vb[c].h[1] = *(const bf16x8*)(vr + 32 * c + 16 + 8 * lg);
      }
#pragma unroll
      for (int v = 0; v < 8; v++) oacc[t][v] *= fr[v];
      oacc[t] = wmma_bf16(pa[0].v, vb[0].v, oacc[t]);
      oacc[t] = wmma_bf16(pa[1].v, vb[1].v, oacc[t]);
    }
  }

  // spill O to LDS once for vectorized, coalesced epilogue
#pragma unroll
  for (int t = 0; t < 4; t++)
#pragma unroll
    for (int v = 0; v < 8; v++)
      Os[w][v + 8 * lg][16 * t + lr] = oacc[t][v];
  __syncthreads();

  // z[b,q, h*VD + d] = O / l  — vectorized b128 stores
  {
    int r = lane >> 1, d0 = (lane & 1) * 32;
    float invl = 1.0f / rowl[w][r];
    size_t rowoff = (size_t)(b * SLN + qb + w * 16 + r) * (NH * VD) + h * VD;
#pragma unroll
    for (int c = 0; c < 4; c++) {
      bf16x8 o;
#pragma unroll
      for (int i = 0; i < 8; i++)
        o[i] = (bf16)(Os[w][r][d0 + c * 8 + i] * invl);
      *(bf16x8*)&Zg[rowoff + d0 + c * 8] = o;
    }
  }
}

// ---------------- fused residual + LayerNorm (unbiased std, ddof=1) ----------
__global__ __launch_bounds__(256) void ln_kernel(
    const float* __restrict__ x, const float* __restrict__ res,
    const float* __restrict__ g, const float* __restrict__ bta,
    float* __restrict__ outf, bf16* __restrict__ outb) {
  __shared__ float s1[256], s2[256];
  const int row = blockIdx.x;
  const int t = threadIdx.x;
  float v0 = x[(size_t)row * ED + t] + (res ? res[(size_t)row * ED + t] : 0.0f);
  float v1 = x[(size_t)row * ED + t + 256] +
             (res ? res[(size_t)row * ED + t + 256] : 0.0f);
  s1[t] = v0 + v1;
  s2[t] = v0 * v0 + v1 * v1;
  __syncthreads();
  for (int off = 128; off > 0; off >>= 1) {
    if (t < off) { s1[t] += s1[t + off]; s2[t] += s2[t + off]; }
    __syncthreads();
  }
  float mean = s1[0] / (float)ED;
  float var  = (s2[0] - (float)ED * mean * mean) / (float)(ED - 1);
  float inv  = 1.0f / (sqrtf(var) + 1e-6f);
  float y0 = g[t] * (v0 - mean) * inv + bta[t];
  float y1 = g[t + 256] * (v1 - mean) * inv + bta[t + 256];
  outf[(size_t)row * ED + t] = y0;
  outf[(size_t)row * ED + t + 256] = y1;
  outb[(size_t)row * ED + t] = (bf16)y0;
  outb[(size_t)row * ED + t + 256] = (bf16)y1;
}

// ---------------- driver ----------------
extern "C" void kernel_launch(void* const* d_in, const int* in_sizes, int n_in,
                              void* d_out, int out_size, void* d_ws,
                              size_t ws_size, hipStream_t stream) {
  (void)in_sizes; (void)n_in; (void)out_size; (void)ws_size;
  const float* src = (const float*)d_in[0];
  const int*   msk = (const int*)d_in[1];
  const float* WQw = (const float*)d_in[2];  const float* WQb = (const float*)d_in[3];
  const float* WKw = (const float*)d_in[4];  const float* WKb = (const float*)d_in[5];
  const float* WVw = (const float*)d_in[6];  const float* WVb = (const float*)d_in[7];
  const float* WOw = (const float*)d_in[8];  const float* WOb = (const float*)d_in[9];
  const float* F1w = (const float*)d_in[10]; const float* F1b = (const float*)d_in[11];
  const float* F2w = (const float*)d_in[12]; const float* F2b = (const float*)d_in[13];
  const float* lng = (const float*)d_in[14]; const float* lnb = (const float*)d_in[15];

  char* p = (char*)d_ws;
  auto alloc = [&](size_t bytes) -> void* {
    void* r = (void*)p;
    p += (bytes + 255) & ~(size_t)255;
    return r;
  };
  bf16* wq_bf  = (bf16*)alloc((size_t)LAY * ED * ED * 2);   // transposed N x K
  bf16* wk_bf  = (bf16*)alloc((size_t)LAY * ED * ED * 2);
  bf16* wv_bf  = (bf16*)alloc((size_t)LAY * ED * ED * 2);
  bf16* wo_bf  = (bf16*)alloc((size_t)LAY * ED * ED * 2);
  bf16* f1_bf  = (bf16*)alloc((size_t)LAY * ED * FF * 2);
  bf16* f2_bf  = (bf16*)alloc((size_t)LAY * FF * ED * 2);
  bf16* x_bf   = (bf16*)alloc((size_t)NTOK * ED * 2);
  bf16* q_bf   = (bf16*)alloc((size_t)NTOK * ED * 2);
  bf16* k_bf   = (bf16*)alloc((size_t)NTOK * ED * 2);
  bf16* v_bf   = (bf16*)alloc((size_t)NTOK * ED * 2);
  bf16* z_bf   = (bf16*)alloc((size_t)NTOK * ED * 2);
  bf16* t_bf   = (bf16*)alloc((size_t)NTOK * ED * 2);
  bf16* ffh_bf = (bf16*)alloc((size_t)NTOK * FF * 2);
  float* attn_f32 = (float*)alloc((size_t)NTOK * ED * 4);
  float* t_f32    = (float*)alloc((size_t)NTOK * ED * 4);
  float* ff2_f32  = (float*)alloc((size_t)NTOK * ED * 4);
  float* xbuf_f32 = (float*)alloc((size_t)NTOK * ED * 4);

  auto cvt = [&](const float* in, bf16* out, size_t n) {
    cvt_f32_bf16<<<dim3((unsigned)((n + 255) / 256)), 256, 0, stream>>>(
        in, out, (int)n);
  };
  auto cvtT = [&](const float* in, bf16* out, int K, int N) {
    for (int l = 0; l < LAY; l++)
      cvt_t<<<dim3((unsigned)(((size_t)K * N + 255) / 256)), 256, 0, stream>>>(
          in + (size_t)l * K * N, out + (size_t)l * N * K, K, N);
  };
  auto gemm = [&](const bf16* A, const bf16* B, const float* bias, float* Cf,
                  bf16* Cb, int N, int K, bool relu) {
    dim3 grid(N / 128, NTOK / 128);
    if (relu)
      gemm_bf16<1><<<grid, 256, 0, stream>>>(A, B, bias, Cf, Cb, NTOK, N, K);
    else
      gemm_bf16<0><<<grid, 256, 0, stream>>>(A, B, bias, Cf, Cb, NTOK, N, K);
  };

  cvtT(WQw, wq_bf, ED, ED);
  cvtT(WKw, wk_bf, ED, ED);
  cvtT(WVw, wv_bf, ED, ED);
  cvtT(WOw, wo_bf, ED, ED);
  cvtT(F1w, f1_bf, ED, FF);
  cvtT(F2w, f2_bf, FF, ED);

  const float* x_f32 = src;
  cvt(src, x_bf, (size_t)NTOK * ED);

  for (int l = 0; l < LAY; l++) {
    const bf16* wq = wq_bf + (size_t)l * ED * ED;
    const bf16* wk = wk_bf + (size_t)l * ED * ED;
    const bf16* wv = wv_bf + (size_t)l * ED * ED;
    const bf16* wo = wo_bf + (size_t)l * ED * ED;
    const bf16* f1 = f1_bf + (size_t)l * ED * FF;
    const bf16* f2 = f2_bf + (size_t)l * FF * ED;

    gemm(x_bf, wq, WQb + (size_t)l * ED, nullptr, q_bf, ED, ED, false);
    gemm(x_bf, wk, WKb + (size_t)l * ED, nullptr, k_bf, ED, ED, false);
    gemm(x_bf, wv, WVb + (size_t)l * ED, nullptr, v_bf, ED, ED, false);

    attention_kernel<<<dim3(SLN / 64, NH, BSZ), 128, 0, stream>>>(
        q_bf, k_bf, v_bf, msk, z_bf);

    gemm(z_bf, wo, WOb + (size_t)l * ED, attn_f32, nullptr, ED, ED, false);
    ln_kernel<<<NTOK, 256, 0, stream>>>(attn_f32, x_f32,
                                        lng + (size_t)l * ED,
                                        lnb + (size_t)l * ED, t_f32, t_bf);

    gemm(t_bf, f1, F1b + (size_t)l * FF, nullptr, ffh_bf, FF, ED, true);
    gemm(ffh_bf, f2, F2b + (size_t)l * ED, ff2_f32, nullptr, ED, FF, false);

    float* xo = (l == LAY - 1) ? (float*)d_out : xbuf_f32;
    ln_kernel<<<NTOK, 256, 0, stream>>>(ff2_f32, t_f32,
                                        lng + (size_t)l * ED,
                                        lnb + (size_t)l * ED, xo, x_bf);
    x_f32 = xbuf_f32;
  }
}
